// CustomGATLayerEff_35021163331732
// MI455X (gfx1250) — matrix operand order repaired and checked by reading, not compiled, not run on hardware
//
#include <hip/hip_runtime.h>

typedef __attribute__((ext_vector_type(16))) __bf16 v16bf;
typedef __attribute__((ext_vector_type(8)))  float  v8f;

#define N     8192
#define DIN   256
#define DOUT  64
#define LOG2E 1.4426950408889634f
#define ALPHA 0.2f

// ---------------------------------------------------------------------------
// Kernel 1: Wh = inputs @ W ; WhT (bf16, [DOUT][N]) ; Wh1s/Wh2s = (Wh@a)*log2e
// One block per row, 64 threads (one per output dim). Tiny fraction of work.
// ---------------------------------------------------------------------------
__global__ __launch_bounds__(64) void gat_prep(
    const float* __restrict__ inputs, const float* __restrict__ W,
    const float* __restrict__ a, __bf16* __restrict__ whT,
    float* __restrict__ wh1s, float* __restrict__ wh2s) {
  const int row = blockIdx.x;
  const int d   = threadIdx.x;                 // 0..63
  const float* inr = inputs + (size_t)row * DIN;
  float acc = 0.f;
#pragma unroll 8
  for (int k = 0; k < DIN; ++k)
    acc = fmaf(inr[k], W[k * DOUT + d], acc);
  // transposed bf16 copy for WMMA B operand (contiguous K per output dim)
  whT[(size_t)d * N + row] = (__bf16)acc;
  // block reduction for Wh1 = Wh@a[:64], Wh2 = Wh@a[64:], pre-scaled by log2e
  __shared__ float s1[64], s2[64];
  s1[d] = acc * a[d];
  s2[d] = acc * a[DOUT + d];
  __syncthreads();
  for (int off = 32; off > 0; off >>= 1) {
    if (d < off) { s1[d] += s1[d + off]; s2[d] += s2[d + off]; }
    __syncthreads();
  }
  if (d == 0) { wh1s[row] = s1[0] * LOG2E; wh2s[row] = s2[0] * LOG2E; }
}

// ---------------------------------------------------------------------------
// Kernel 2: fused masked-softmax attention * Wh (flash-attention style).
// One BLOCK (4 waves) per 16-row tile; the 8192-column loop is strided across
// the 4 waves (j0 = wave*32 + m*128) for 2048 independent waves chip-wide.
// P is computed directly as the 16x16x32 bf16 WMMA A fragment, in this code's contiguous K order (B matches):
//   lane <16 : M=lane,    K in {0..15}
//   lane>=16 : M=lane-16, K in {16..31}
// Partial 16x64 accumulators + row sums are combined via LDS ds_add_f32.
// ---------------------------------------------------------------------------
__global__ __launch_bounds__(128) void gat_attn(
    const int* __restrict__ adj, const __bf16* __restrict__ whT,
    const float* __restrict__ wh1s, const float* __restrict__ wh2s,
    float* __restrict__ out) {
  const int lane    = threadIdx.x & 31;
  const int wave    = threadIdx.x >> 5;        // 0..3
  const int i0      = blockIdx.x * 16;         // this block's row tile
  const int halfsel = lane >> 4;               // 0 or 1
  const int rloc    = lane & 15;
  const int row     = i0 + rloc;               // row this lane builds P for
  const int laneoff = halfsel * 16;

  const float wh1r = wh1s[row];
  const int* adjrow = adj + (size_t)row * N;
  const int jdiag = i0 & ~31;                  // column tile containing diag

  __shared__ float accsh[16 * DOUT];           // 4 KB combined output tile
  __shared__ float lsh[16];                    // per-row softmax denominators
  {
    const int tid = threadIdx.x;
#pragma unroll
    for (int k = 0; k < 8; ++k) accsh[tid * 8 + k] = 0.f;
    if (tid < 16) lsh[tid] = 0.f;
  }
  __syncthreads();

  v8f c[4]; c[0] = (v8f){}; c[1] = (v8f){}; c[2] = (v8f){}; c[3] = (v8f){};
  float lsum = 0.f;

  for (int j0 = wave * 32; j0 < N; j0 += 128) {
    const int cb = j0 + laneoff;
    // adj + wh2 for this lane's 16 columns: cb+{0..15}
    const int4*   ap = (const int4*)(adjrow + cb);
    const float4* wp = (const float4*)(wh2s + cb);
    int4   A0 = ap[0], A1 = ap[1], A2 = ap[2], A3 = ap[3];
    float4 W0 = wp[0], W1 = wp[1], W2 = wp[2], W3 = wp[3];
    // prefetch this wave's next adj tile
    __builtin_prefetch(adjrow + cb + 128, 0, 0);

    const int   av[16] = {A0.x, A0.y, A0.z, A0.w, A1.x, A1.y, A1.z, A1.w,
                          A2.x, A2.y, A2.z, A2.w, A3.x, A3.y, A3.z, A3.w};
    const float wv[16] = {W0.x, W0.y, W0.z, W0.w, W1.x, W1.y, W1.z, W1.w,
                          W2.x, W2.y, W2.z, W2.w, W3.x, W3.y, W3.z, W3.w};

    const bool dtile = (j0 == jdiag);
    float p[16];
    v16bf afrag;
#pragma unroll
    for (int k = 0; k < 16; ++k) {
      const int col = cb + k;
      float t = wh1r + wv[k];                      // (Wh1+Wh2)*log2e
      t = fmaxf(t, ALPHA * t);                     // leaky (scale-invariant)
      float pe = __builtin_amdgcn_exp2f(t);        // exp(e) via v_exp_f32
      bool keep = (av[k] != 0);
      if (dtile) keep = keep || (col == row);      // self loop (adj + I)
      p[k] = keep ? pe : 0.f;
      afrag[k] = (__bf16)p[k];
    }
    // log-depth tree sum (short critical path, dual-issue friendly)
#pragma unroll
    for (int s = 1; s < 16; s <<= 1)
#pragma unroll
      for (int k = 0; k < 16; k += 2 * s) p[k] += p[k + s];
    lsum += p[0];

    // B fragments: WhT[d][j] bf16, 32B contiguous per lane, WMMA B layout
    const __bf16* bbase =
        whT + (size_t)rloc * N + (size_t)(j0 + halfsel * 16);
#pragma unroll
    for (int t = 0; t < 4; ++t) {
      v16bf bfrag = *(const v16bf*)(bbase + (size_t)t * 16 * N);
      c[t] = __builtin_amdgcn_wmma_f32_16x16x32_bf16(
          false, afrag, false, bfrag, (short)0, c[t], false, false);
    }
  }

  // combine the 4 waves' partial results in LDS (ds_add_f32 atomics).
  // lsum: lanes L and L^16 hold disjoint K-halves of row rloc -> just add both.
  atomicAdd(&lsh[rloc], lsum);
#pragma unroll
  for (int r = 0; r < 8; ++r) {
    const int rr = r + halfsel * 8;              // C layout: lanes>=16 -> M+8
#pragma unroll
    for (int t = 0; t < 4; ++t)
      atomicAdd(&accsh[rr * DOUT + t * 16 + rloc], c[t][r]);
  }
  __syncthreads();

  // normalize + store: 1024 outputs over 128 threads, 8 contiguous each
  {
    const int base = threadIdx.x * 8;            // within 16x64 tile
    const int r    = base >> 6;                  // row in tile
    const float inv = __builtin_amdgcn_rcpf(lsh[r]);
    float* orow = out + (size_t)(i0 + r) * DOUT + (base & 63);
#pragma unroll
    for (int k = 0; k < 8; ++k) orow[k] = accsh[base + k] * inv;
  }
}

extern "C" void kernel_launch(void* const* d_in, const int* in_sizes, int n_in,
                              void* d_out, int out_size, void* d_ws, size_t ws_size,
                              hipStream_t stream) {
  const int*   adj    = (const int*)d_in[0];
  const float* inputs = (const float*)d_in[1];
  const float* W      = (const float*)d_in[2];
  const float* a      = (const float*)d_in[3];
  float*       out    = (float*)d_out;

  char* ws = (char*)d_ws;
  __bf16* whT  = (__bf16*)ws;                              // 64*8192*2 = 1 MB
  float*  wh1s = (float*)(ws + (size_t)DOUT * N * 2);      // 32 KB
  float*  wh2s = wh1s + N;                                 // 32 KB

  gat_prep<<<N, 64, 0, stream>>>(inputs, W, a, whT, wh1s, wh2s);
  gat_attn<<<N / 16, 128, 0, stream>>>(adj, whT, wh1s, wh2s, out);
}
